// CustomGRU_21500606284385
// MI455X (gfx1250) — matrix operand order repaired and checked by reading, not compiled, not run on hardware
//
#include <hip/hip_runtime.h>

// ---------------- problem constants ----------------
constexpr int Bq = 16;
constexpr int Tq = 512;
constexpr int EMBq = 512;
constexpr int HIDq = 1024;
constexpr int H3q = 3 * HIDq;       // 3072
constexpr int VOCq = 32000;
constexpr int Mrows = Bq * Tq;      // 8192, row r = t*16 + b  (time-major)
constexpr size_t LOGITS_ELEMS = (size_t)Bq * Tq * VOCq;   // 262,144,000

// ---------------- WMMA fragment types ----------------
typedef __attribute__((ext_vector_type(16))) __bf16 v16bf;
typedef __attribute__((ext_vector_type(8)))  float  v8f;

union FragU { v16bf v; uint4 u[2]; };

// A fragment: 16x32 bf16, row-major source, stride lda (elements).
// lane&15 = M row; lane>=16 selects K {8..15,24..31} instead of {0..7,16..23}.
__device__ __forceinline__ v16bf load_frag_a(const __bf16* __restrict__ base, int lda, int lane) {
  const int m   = lane & 15;
  const int sel = (lane >> 4) & 1;
  const __bf16* p = base + (size_t)m * lda + sel * 8;
  FragU f;
  f.u[0] = *(const uint4*)(p);
  f.u[1] = *(const uint4*)(p + 16);
  return f.v;
}

// B fragment: 32x16 bf16 sourced from row-major W[N,K] (i.e. Bᵀ), stride ldb.
// lane&15 = N col; lane>=16 selects K {8..15,24..31} instead of {0..7,16..23}.
__device__ __forceinline__ v16bf load_frag_b(const __bf16* __restrict__ base, int ldb, int lane) {
  const int n    = lane & 15;
  const int koff = (lane & 16) >> 1;
  const __bf16* p = base + (size_t)n * ldb + koff;
  FragU f;
  f.u[0] = *(const uint4*)(p);
  f.u[1] = *(const uint4*)(p + 16);
  return f.v;
}

__device__ __forceinline__ float sigmoidf_(float x) {
  return 1.0f / (1.0f + __expf(-x));
}

// Issue one async global->LDS 128b copy (GV addressing, tracked by ASYNCcnt).
__device__ __forceinline__ void async_b128(unsigned lds_byte_addr, const void* gaddr) {
  asm volatile("global_load_async_to_lds_b128 %0, %1, off"
               :: "v"(lds_byte_addr), "v"(gaddr)
               : "memory");
}

// Low 32 bits of a generic shared-memory address == workgroup LDS byte offset
// (ISA 10.2: LDS aperture maps addr[31:0] to LDS).
__device__ __forceinline__ unsigned lds_off(const void* p) {
  return (unsigned)(uintptr_t)p;
}

// ---------------- big WMMA GEMM: C[M,N] = A[M,K](bf16) * W[N,K]ᵀ(bf16) + bias ----------------
// Block 256 thr = 8 waves (2 M-waves x 4 N-waves). Block tile 64M x 256N,
// wave tile 32M x 64N (2 A-frags x 4 B-frags -> 8 WMMAs / k-step).
// K staged in 32-wide chunks through double-buffered LDS via async copies
// (5 b128 per thread per chunk, drained with s_wait_asynccnt + barrier).
// LDS rows padded to 40 halves (stride 20 banks, conflict-free b128 frag loads).
// Last K-chunk peeled so the steady-state loop body is branch-free.
// permute_tb: output row r=t*16+b stored at (b*Tq + t) for [B,T,V] logits.
__global__ __launch_bounds__(256)
void wmma_gemm_bias(const __bf16* __restrict__ A, const __bf16* __restrict__ W,
                    const float* __restrict__ bias, float* __restrict__ C,
                    int K, int N, int permute_tb) {
  __shared__ __bf16 sA[2][64][40];    //  10,240 B
  __shared__ __bf16 sB[2][256][40];   //  40,960 B

  const int tid  = threadIdx.x;
  const int lane = tid & 31;
  const int wave = tid >> 5;
  const int wm = wave & 1;            // 0..1  (32 rows each)
  const int wn = wave >> 1;           // 0..3  (64 cols each)

  // Per-thread async source pointers, advanced 64B (32 halves) per chunk.
  const int ar = tid >> 2;            // 0..63 : A row
  const int aq = tid & 3;             // 16B quarter of the 64B row chunk
  const char* agp = (const char*)(A + ((size_t)(blockIdx.y * 64) + ar) * K) + aq * 16;
  const char* bgp = (const char*)(W + ((size_t)(blockIdx.x * 256) + tid) * K);

  auto issue_chunk = [&](int buf) {
    async_b128(lds_off(&sA[buf][ar][aq * 8]), agp);
    const unsigned bl = lds_off(&sB[buf][tid][0]);
#pragma unroll
    for (int q2 = 0; q2 < 4; ++q2)
      async_b128(bl + q2 * 16, bgp + q2 * 16);
    agp += 64;
    bgp += 64;
  };

  v8f acc[2][4] = {};

  auto compute_chunk = [&](int buf) {
    const __bf16* sa = &sA[buf][wm * 32][0];
    const __bf16* sb = &sB[buf][wn * 64][0];
    // Load all fragments first so ds_load_b128s clause together.
    v16bf a0 = load_frag_a(sa,           40, lane);
    v16bf a1 = load_frag_a(sa + 16 * 40, 40, lane);
    v16bf b0 = load_frag_b(sb +  0 * 40, 40, lane);
    v16bf b1 = load_frag_b(sb + 16 * 40, 40, lane);
    v16bf b2 = load_frag_b(sb + 32 * 40, 40, lane);
    v16bf b3 = load_frag_b(sb + 48 * 40, 40, lane);
    acc[0][0] = __builtin_amdgcn_wmma_f32_16x16x32_bf16(false, a0, false, b0, (short)0, acc[0][0], false, false);
    acc[1][0] = __builtin_amdgcn_wmma_f32_16x16x32_bf16(false, a1, false, b0, (short)0, acc[1][0], false, false);
    acc[0][1] = __builtin_amdgcn_wmma_f32_16x16x32_bf16(false, a0, false, b1, (short)0, acc[0][1], false, false);
    acc[1][1] = __builtin_amdgcn_wmma_f32_16x16x32_bf16(false, a1, false, b1, (short)0, acc[1][1], false, false);
    acc[0][2] = __builtin_amdgcn_wmma_f32_16x16x32_bf16(false, a0, false, b2, (short)0, acc[0][2], false, false);
    acc[1][2] = __builtin_amdgcn_wmma_f32_16x16x32_bf16(false, a1, false, b2, (short)0, acc[1][2], false, false);
    acc[0][3] = __builtin_amdgcn_wmma_f32_16x16x32_bf16(false, a0, false, b3, (short)0, acc[0][3], false, false);
    acc[1][3] = __builtin_amdgcn_wmma_f32_16x16x32_bf16(false, a1, false, b3, (short)0, acc[1][3], false, false);
  };

  issue_chunk(0);
  int cur = 0;
  // steady state: branch-free body, prefetch next chunk while computing current
  for (int k = 0; k < K - 32; k += 32) {
    issue_chunk(cur ^ 1);
    asm volatile("s_wait_asynccnt 0x5" ::: "memory");   // prev buffer drained
    __syncthreads();
    compute_chunk(cur);
    __syncthreads();                                    // protect buffer reuse
    cur ^= 1;
  }
  // peeled last chunk
  asm volatile("s_wait_asynccnt 0x0" ::: "memory");
  __syncthreads();
  compute_chunk(cur);

  const int cl = lane & 15;
#pragma unroll
  for (int f = 0; f < 2; ++f) {
    const int rb = blockIdx.y * 64 + wm * 32 + f * 16 + ((lane >> 4) << 3);
#pragma unroll
    for (int j = 0; j < 4; ++j) {
      const int col = blockIdx.x * 256 + wn * 64 + j * 16 + cl;
      const float bv = bias[col];
#pragma unroll
      for (int g = 0; g < 8; ++g) {
        const int r = rb + g;
        const size_t ro = permute_tb ? (size_t)((r & 15) * Tq + (r >> 4)) : (size_t)r;
        C[ro * (size_t)N + col] = acc[f][j][g] + bv;
      }
    }
  }
}

// ---------------- GRU layer 0 step: gh0 = h0 @ Whh0ᵀ, fused gates ----------------
// 64 blocks x 1 wave; block handles gate columns [j0, j0+16) across r/u/n thirds.
__global__ __launch_bounds__(32)
void gru0_step(const float* __restrict__ gi_t,      // [16 x 3072] (x@Wih0ᵀ + b_ih0)
               const __bf16* __restrict__ hb_in,    // [16 x 1024]
               const float* __restrict__ hf_in,     // [16 x 1024]
               const __bf16* __restrict__ Whh,      // [3072 x 1024] bf16
               const float* __restrict__ bhh,       // [3072]
               float* __restrict__ hf_out,
               __bf16* __restrict__ hb_out) {
  const int lane = threadIdx.x;
  const int j0 = blockIdx.x * 16;

  v8f aR = {}, aU = {}, aN = {};
  const __bf16* Wr = Whh + (size_t)(j0            ) * HIDq;
  const __bf16* Wu = Whh + (size_t)(j0 +     HIDq ) * HIDq;
  const __bf16* Wn = Whh + (size_t)(j0 + 2 * HIDq ) * HIDq;
  for (int k = 0; k < HIDq; k += 32) {
    v16bf a  = load_frag_a(hb_in + k, HIDq, lane);
    v16bf br = load_frag_b(Wr + k, HIDq, lane);
    v16bf bu = load_frag_b(Wu + k, HIDq, lane);
    v16bf bn = load_frag_b(Wn + k, HIDq, lane);
    aR = __builtin_amdgcn_wmma_f32_16x16x32_bf16(false, a, false, br, (short)0, aR, false, false);
    aU = __builtin_amdgcn_wmma_f32_16x16x32_bf16(false, a, false, bu, (short)0, aU, false, false);
    aN = __builtin_amdgcn_wmma_f32_16x16x32_bf16(false, a, false, bn, (short)0, aN, false, false);
  }

  const int col = j0 + (lane & 15);
  const float br_ = bhh[col], bu_ = bhh[HIDq + col], bn_ = bhh[2 * HIDq + col];
  const int rb = (lane >> 4) << 3;
#pragma unroll
  for (int g = 0; g < 8; ++g) {
    const int m = rb + g;
    const float gir = gi_t[m * H3q + col];
    const float giu = gi_t[m * H3q + HIDq + col];
    const float gin = gi_t[m * H3q + 2 * HIDq + col];
    const float r = sigmoidf_(gir + aR[g] + br_);
    const float u = sigmoidf_(giu + aU[g] + bu_);
    const float n = tanhf(gin + r * (aN[g] + bn_));
    const float hnew = (1.0f - u) * n + u * hf_in[m * HIDq + col];
    hf_out[m * HIDq + col] = hnew;
    hb_out[m * HIDq + col] = (__bf16)hnew;
  }
}

// ---------------- GRU layer 1 step: gi1 = h0ₜ @ Wih1ᵀ, gh1 = h1 @ Whh1ᵀ, fused gates ----------------
__global__ __launch_bounds__(32)
void gru1_step(const __bf16* __restrict__ xb,       // h0_new [16 x 1024] bf16
               const __bf16* __restrict__ hb_in,    // h1 [16 x 1024] bf16
               const float* __restrict__ hf_in,     // h1 [16 x 1024] f32
               const __bf16* __restrict__ Wih,      // [3072 x 1024]
               const float* __restrict__ bih,
               const __bf16* __restrict__ Whh,      // [3072 x 1024]
               const float* __restrict__ bhh,
               float* __restrict__ hf_out,
               __bf16* __restrict__ hb_out,
               __bf16* __restrict__ hseq_t) {       // [16 x 1024] bf16 slice of Hseq
  const int lane = threadIdx.x;
  const int j0 = blockIdx.x * 16;

  v8f gR = {}, gU = {}, gN = {}, hR = {}, hU = {}, hN = {};
  const __bf16* WiR = Wih + (size_t)(j0            ) * HIDq;
  const __bf16* WiU = Wih + (size_t)(j0 +     HIDq ) * HIDq;
  const __bf16* WiN = Wih + (size_t)(j0 + 2 * HIDq ) * HIDq;
  const __bf16* WhR = Whh + (size_t)(j0            ) * HIDq;
  const __bf16* WhU = Whh + (size_t)(j0 +     HIDq ) * HIDq;
  const __bf16* WhN = Whh + (size_t)(j0 + 2 * HIDq ) * HIDq;
  for (int k = 0; k < HIDq; k += 32) {
    v16bf ax = load_frag_a(xb + k, HIDq, lane);
    v16bf ah = load_frag_a(hb_in + k, HIDq, lane);
    v16bf b;
    b = load_frag_b(WiR + k, HIDq, lane);
    gR = __builtin_amdgcn_wmma_f32_16x16x32_bf16(false, ax, false, b, (short)0, gR, false, false);
    b = load_frag_b(WiU + k, HIDq, lane);
    gU = __builtin_amdgcn_wmma_f32_16x16x32_bf16(false, ax, false, b, (short)0, gU, false, false);
    b = load_frag_b(WiN + k, HIDq, lane);
    gN = __builtin_amdgcn_wmma_f32_16x16x32_bf16(false, ax, false, b, (short)0, gN, false, false);
    b = load_frag_b(WhR + k, HIDq, lane);
    hR = __builtin_amdgcn_wmma_f32_16x16x32_bf16(false, ah, false, b, (short)0, hR, false, false);
    b = load_frag_b(WhU + k, HIDq, lane);
    hU = __builtin_amdgcn_wmma_f32_16x16x32_bf16(false, ah, false, b, (short)0, hU, false, false);
    b = load_frag_b(WhN + k, HIDq, lane);
    hN = __builtin_amdgcn_wmma_f32_16x16x32_bf16(false, ah, false, b, (short)0, hN, false, false);
  }

  const int col = j0 + (lane & 15);
  const float bir = bih[col], biu = bih[HIDq + col], bin = bih[2 * HIDq + col];
  const float bhr = bhh[col], bhu = bhh[HIDq + col], bhn = bhh[2 * HIDq + col];
  const int rb = (lane >> 4) << 3;
#pragma unroll
  for (int g = 0; g < 8; ++g) {
    const int m = rb + g;
    const float r = sigmoidf_((gR[g] + bir) + (hR[g] + bhr));
    const float u = sigmoidf_((gU[g] + biu) + (hU[g] + bhu));
    const float n = tanhf((gN[g] + bin) + r * (hN[g] + bhn));
    const float hnew = (1.0f - u) * n + u * hf_in[m * HIDq + col];
    hf_out[m * HIDq + col] = hnew;
    hb_out[m * HIDq + col] = (__bf16)hnew;
    hseq_t[m * HIDq + col] = (__bf16)hnew;
  }
}

// ---------------- helpers ----------------
__global__ void cast_f32_bf16(const float* __restrict__ in, __bf16* __restrict__ out, int n) {
  int i = blockIdx.x * blockDim.x + threadIdx.x;
  if (i < n) out[i] = (__bf16)in[i];
}

// Gather embeddings to time-major bf16 X[t*16+b, 0..511]
__global__ void embed_gather(const int* __restrict__ seq, const float* __restrict__ emb,
                             __bf16* __restrict__ Xb) {
  const int r = blockIdx.x;               // 0..8191 = t*16 + b
  const int t = r >> 4, b = r & 15;
  const int tok = seq[b * Tq + t];
  const float* e = emb + (size_t)tok * EMBq;
  __bf16* o = Xb + (size_t)r * EMBq;
  for (int i = threadIdx.x; i < EMBq; i += blockDim.x) o[i] = (__bf16)e[i];
}

__global__ void zero_h(float* h0f, __bf16* h0b, float* h1f, __bf16* h1b) {
  int i = blockIdx.x * blockDim.x + threadIdx.x;
  if (i < Bq * HIDq) {
    h0f[i] = 0.0f; h0b[i] = (__bf16)0.0f;
    h1f[i] = 0.0f; h1b[i] = (__bf16)0.0f;
  }
}

__global__ void finalize_h(const float* __restrict__ h0, const float* __restrict__ h1,
                           float* __restrict__ out) {
  int i = blockIdx.x * blockDim.x + threadIdx.x;
  if (i < Bq * HIDq) {
    out[LOGITS_ELEMS + i] = h0[i];
    out[LOGITS_ELEMS + (size_t)(Bq * HIDq) + i] = h1[i];
  }
}

// ---------------- host launcher ----------------
extern "C" void kernel_launch(void* const* d_in, const int* in_sizes, int n_in,
                              void* d_out, int out_size, void* d_ws, size_t ws_size,
                              hipStream_t stream) {
  const int*   seq    = (const int*)  d_in[0];
  const float* emb    = (const float*)d_in[1];
  const float* w_ih_0 = (const float*)d_in[2];
  const float* w_hh_0 = (const float*)d_in[3];
  const float* b_ih_0 = (const float*)d_in[4];
  const float* b_hh_0 = (const float*)d_in[5];
  const float* w_ih_1 = (const float*)d_in[6];
  const float* w_hh_1 = (const float*)d_in[7];
  const float* b_ih_1 = (const float*)d_in[8];
  const float* b_hh_1 = (const float*)d_in[9];
  const float* w_out  = (const float*)d_in[10];
  const float* b_out  = (const float*)d_in[11];
  float* out = (float*)d_out;

  // --- carve workspace (256B aligned) ---
  char* base = (char*)d_ws;
  size_t off = 0;
  auto carve = [&](size_t bytes) -> char* {
    char* p = base + off;
    off = (off + bytes + 255) & ~(size_t)255;
    return p;
  };
  __bf16* Xb    = (__bf16*)carve((size_t)Mrows * EMBq * 2);          // 8 MB
  __bf16* Wih0b = (__bf16*)carve((size_t)H3q * EMBq * 2);            // 3 MB
  __bf16* Whh0b = (__bf16*)carve((size_t)H3q * HIDq * 2);            // 6 MB
  __bf16* Wih1b = (__bf16*)carve((size_t)H3q * HIDq * 2);            // 6 MB
  __bf16* Whh1b = (__bf16*)carve((size_t)H3q * HIDq * 2);            // 6 MB
  __bf16* Woutb = (__bf16*)carve((size_t)VOCq * HIDq * 2);           // 64 MB
  float*  GI0   = (float*) carve((size_t)Mrows * H3q * 4);           // 96 MB
  __bf16* Hseq  = (__bf16*)carve((size_t)Mrows * HIDq * 2);          // 16 MB
  float*  h0f[2] = {(float*)carve(Bq * HIDq * 4), (float*)carve(Bq * HIDq * 4)};
  __bf16* h0b[2] = {(__bf16*)carve(Bq * HIDq * 2), (__bf16*)carve(Bq * HIDq * 2)};
  float*  h1f[2] = {(float*)carve(Bq * HIDq * 4), (float*)carve(Bq * HIDq * 4)};
  __bf16* h1b[2] = {(__bf16*)carve(Bq * HIDq * 2), (__bf16*)carve(Bq * HIDq * 2)};

  // --- 1. bf16 conversion of weights + embedding gather ---
  auto cast = [&](const float* src, __bf16* dst, int n) {
    cast_f32_bf16<<<(n + 255) / 256, 256, 0, stream>>>(src, dst, n);
  };
  cast(w_ih_0, Wih0b, H3q * EMBq);
  cast(w_hh_0, Whh0b, H3q * HIDq);
  cast(w_ih_1, Wih1b, H3q * HIDq);
  cast(w_hh_1, Whh1b, H3q * HIDq);
  cast(w_out,  Woutb, VOCq * HIDq);
  embed_gather<<<Mrows, 128, 0, stream>>>(seq, emb, Xb);
  zero_h<<<(Bq * HIDq + 255) / 256, 256, 0, stream>>>(h0f[0], h0b[0], h1f[0], h1b[0]);

  // --- 2. hoisted input GEMM: GI0[8192,3072] = Xb @ Wih0ᵀ + b_ih0 ---
  wmma_gemm_bias<<<dim3(H3q / 256, Mrows / 64), 256, 0, stream>>>(
      Xb, Wih0b, b_ih_0, GI0, EMBq, H3q, 0);

  // --- 3. sequential recurrence: 2 fused WMMA kernels per timestep ---
  for (int t = 0; t < Tq; ++t) {
    const int pi = t & 1, po = (t + 1) & 1;
    gru0_step<<<HIDq / 16, 32, 0, stream>>>(
        GI0 + (size_t)t * Bq * H3q, h0b[pi], h0f[pi], Whh0b, b_hh_0, h0f[po], h0b[po]);
    gru1_step<<<HIDq / 16, 32, 0, stream>>>(
        h0b[po], h1b[pi], h1f[pi], Wih1b, b_ih_1, Whh1b, b_hh_1,
        h1f[po], h1b[po], Hseq + (size_t)t * Bq * HIDq);
  }

  // --- 4. output projection: logits[B,T,V] = Hseq @ w_outᵀ + b_out (permuted store) ---
  wmma_gemm_bias<<<dim3(VOCq / 256, Mrows / 64), 256, 0, stream>>>(
      Hseq, Woutb, b_out, out, HIDq, VOCq, 1);

  // --- 5. final hidden states (ping-pong index 0 after 512 steps) ---
  finalize_h<<<(Bq * HIDq + 255) / 256, 256, 0, stream>>>(h0f[0], h1f[0], out);
}
